// PointNetSetAbstraction_6270652252985
// MI455X (gfx1250) — matrix operand order repaired and checked by reading, not compiled, not run on hardware
//
#include <hip/hip_runtime.h>

// ---------- types ----------
typedef _Float16 h16;
typedef _Float16 v16h __attribute__((ext_vector_type(16)));
typedef _Float16 v8h  __attribute__((ext_vector_type(8)));
typedef float    v8f  __attribute__((ext_vector_type(8)));

// ---------- problem constants ----------
#define BB      16
#define NN      4096
#define DD      64
#define NPT     1024
#define NSAMP   32
#define RAD2    0.04f            // 0.2^2
#define RROWS   (BB*NPT*NSAMP)   // 524288 rows of the big GEMMs
#define BN_EPS  1e-5f

// =====================================================================
// 1) Farthest point sampling: one workgroup per batch, xyz in LDS,
//    per-thread running min-distances in registers. Argmax via wave32
//    shfl_xor butterflies (no barriers) + 32-entry cross-wave reduce.
// =====================================================================
__global__ __launch_bounds__(1024) void fps_kernel(const float* __restrict__ xyz,
                                                   float* __restrict__ out_newxyz) {
    __shared__ float sx[NN], sy[NN], sz[NN];
    __shared__ float wmaxs[32];
    __shared__ int   wargs[32];
    const int b = blockIdx.x, tid = threadIdx.x;
    const int wv = tid >> 5, lane = tid & 31;
    const float* base = xyz + (size_t)b * NN * 3;
    float dloc[4];
#pragma unroll
    for (int j = 0; j < 4; ++j) {
        int i = tid + j * 1024;
        sx[i] = base[i*3+0]; sy[i] = base[i*3+1]; sz[i] = base[i*3+2];
        dloc[j] = 1e10f;
    }
    __syncthreads();
    int far = 0;
    for (int t = 0; t < NPT; ++t) {
        float cx = sx[far], cy = sy[far], cz = sz[far];
        if (tid == 0) {  // emit centroid BEFORE the distance update (scan semantics)
            float* o = out_newxyz + ((size_t)b * NPT + t) * 3;
            o[0] = cx; o[1] = cy; o[2] = cz;
        }
        float best = -1.0f; int bi = 0;
#pragma unroll
        for (int j = 0; j < 4; ++j) {
            int i = tid + j * 1024;
            float dx = sx[i]-cx, dy = sy[i]-cy, dz = sz[i]-cz;
            float nd = fminf(dloc[j], dx*dx + dy*dy + dz*dz);
            dloc[j] = nd;
            if (nd > best) { best = nd; bi = i; }   // strict > keeps smallest index on tie
        }
        // wave32 butterfly argmax (first-index tie-break)
#pragma unroll
        for (int m = 16; m >= 1; m >>= 1) {
            float ov = __shfl_xor(best, m, 32);
            int   oa = __shfl_xor(bi,   m, 32);
            if (ov > best || (ov == best && oa < bi)) { best = ov; bi = oa; }
        }
        if (lane == 0) { wmaxs[wv] = best; wargs[wv] = bi; }
        __syncthreads();
        if (tid < 32) {
            best = wmaxs[tid]; bi = wargs[tid];
#pragma unroll
            for (int m = 16; m >= 1; m >>= 1) {
                float ov = __shfl_xor(best, m, 32);
                int   oa = __shfl_xor(bi,   m, 32);
                if (ov > best || (ov == best && oa < bi)) { best = ov; bi = oa; }
            }
            if (tid == 0) wargs[0] = bi;
        }
        __syncthreads();
        far = wargs[0];
        __syncthreads();
    }
}

// =====================================================================
// 2) Ball query: wave32 ballot-based ordered append of the first 32
//    indices with dist^2 <= r^2 (== reference sort+truncate+fill).
// =====================================================================
__global__ __launch_bounds__(1024) void ballquery_kernel(const float* __restrict__ xyz,
                                                         const float* __restrict__ newxyz,
                                                         int* __restrict__ idxbuf) {
    __shared__ float sx[NN], sy[NN], sz[NN];
    const int b = blockIdx.x, tid = threadIdx.x;
    const float* base = xyz + (size_t)b * NN * 3;
    for (int i = tid; i < NN; i += 1024) { sx[i]=base[i*3]; sy[i]=base[i*3+1]; sz[i]=base[i*3+2]; }
    __syncthreads();
    const int w = tid >> 5, lane = tid & 31;
    for (int q = 0; q < 32; ++q) {
        const int s = w * 32 + q;
        const float* qp = newxyz + ((size_t)b * NPT + s) * 3;
        const float qx = qp[0], qy = qp[1], qz = qp[2];
        int* op = idxbuf + ((size_t)b * NPT + s) * NSAMP;
        int cnt = 0, firstIdx = 0;
        for (int bse = 0; bse < NN && cnt < NSAMP; bse += 32) {
            const int i = bse + lane;
            float dx = sx[i]-qx, dy = sy[i]-qy, dz = sz[i]-qz;
            bool pred = (dx*dx + dy*dy + dz*dz) <= RAD2;
            unsigned mm = (unsigned)__ballot(pred);        // wave32: 32-bit mask
            if (cnt == 0 && mm) firstIdx = bse + (__ffs(mm) - 1);
            int pos = cnt + __popc(mm & ((1u << lane) - 1u));
            if (pred && pos < NSAMP) op[pos] = i;
            cnt += __popc(mm);
            if (cnt > NSAMP) cnt = NSAMP;
        }
        if (lane >= cnt) op[lane] = firstIdx;              // fill remainder with first hit
    }
}

// =====================================================================
// 3) Gather + concat -> f16 feature matrix, K padded 67 -> 96 (zeros).
//    Row assembled in registers, stored as 16B vectors.
// =====================================================================
__global__ void build_feat_kernel(const float* __restrict__ xyz, const float* __restrict__ points,
                                  const float* __restrict__ newxyz, const int* __restrict__ idxbuf,
                                  h16* __restrict__ featA) {
    int r = blockIdx.x * blockDim.x + threadIdx.x;
    if (r >= RROWS) return;
    const int b = r >> 15;              // / (1024*32)
    const int s = (r >> 5) & 1023;
    const int i = idxbuf[r];
    const float* nx = newxyz + ((size_t)b * NPT + s) * 3;
    const float* px = xyz + ((size_t)b * NN + i) * 3;
    const float* pp = points + ((size_t)b * NN + i) * DD;
    h16 rowbuf[96];
    rowbuf[0] = (h16)(px[0]-nx[0]);
    rowbuf[1] = (h16)(px[1]-nx[1]);
    rowbuf[2] = (h16)(px[2]-nx[2]);
#pragma unroll
    for (int c = 0; c < DD; ++c) rowbuf[3+c] = (h16)pp[c];
#pragma unroll
    for (int c = 3 + DD; c < 96; ++c) rowbuf[c] = (h16)0.f;
    h16* o = featA + (size_t)r * 96;
#pragma unroll
    for (int c = 0; c < 96; c += 8) *(v8h*)(o + c) = *(const v8h*)(rowbuf + c);
}

// Weights f32 (O,Cin) -> f16 padded (O,Cpad)
__global__ void prep_w_kernel(const float* __restrict__ w, h16* __restrict__ wh,
                              int O, int Cin, int Cpad) {
    int t = blockIdx.x * blockDim.x + threadIdx.x;
    if (t >= O * Cpad) return;
    int o = t / Cpad, c = t - o * Cpad;
    wh[t] = (c < Cin) ? (h16)w[o * Cin + c] : (h16)0.f;
}

// =====================================================================
// 4) GEMM via v_wmma_f32_16x16x32_f16.
//    - Weights staged block-wide into LDS with global_load_async_to_lds_b128
//      (ASYNCcnt), then ALL B fragments hoisted into VGPRs once per wave:
//      the row loop runs pure global_load(A) + back-to-back v_wmma.
//    - Grid-stride over 16-row strips; bias folded into accumulator init.
//    Lane layouts per ISA 7.12.2:
//      A: lane<16 row=lane, K={0..7}+{16..23}; lane>=16 K={8..15}+{24..31}
//      B: N = lane&15, K order same as A: lane<16 K={0..7}+{16..23}; lane>=16 K={8..15}+{24..31}
//      D: VGPR j -> M = j + (lane>=16 ? 8 : 0), N = lane&15
// =====================================================================
template<int CT, int KB>   // CT: 16-wide col tiles (O=CT*16); KB: 32-wide k steps (Cpad=KB*32)
__global__ __launch_bounds__(256) void gemm_wmma_kernel(const h16* __restrict__ A,
                                                        const h16* __restrict__ Wh,
                                                        const float* __restrict__ bias,
                                                        float* __restrict__ H, int ldH,
                                                        int rowTiles) {
    constexpr int O    = CT * 16;
    constexpr int Cpad = KB * 32;
    __shared__ __align__(32) h16 sW[O * Cpad];
    const int tid = threadIdx.x;

    // ---- async copy of the whole (O x Cpad) f16 weight block into LDS ----
    {
        const unsigned bytes = (unsigned)(O * Cpad) * 2u;
        const unsigned ldsBase = (unsigned)(size_t)(&sW[0]);
        for (unsigned byteOff = (unsigned)tid * 16u; byteOff < bytes; byteOff += 256u * 16u) {
            unsigned ldsOff = ldsBase + byteOff;
            const void* gp = (const char*)Wh + byteOff;
            asm volatile("global_load_async_to_lds_b128 %0, %1, off"
                         :: "v"(ldsOff), "v"(gp) : "memory");
        }
        asm volatile("s_wait_asynccnt 0x0" ::: "memory");
    }
    __syncthreads();

    const int wave = tid >> 5;
    const int lane = tid & 31;
    const int hi = lane >> 4;      // upper half-wave
    const int lr = lane & 15;

    // ---- hoist all B fragments + bias into registers (reused for every strip) ----
    v16h bfrag[CT][KB];
    float bv[CT];
#pragma unroll
    for (int ct = 0; ct < CT; ++ct) {
        bv[ct] = bias[ct * 16 + lr];
#pragma unroll
        for (int kb = 0; kb < KB; ++kb)
            bfrag[ct][kb] = __builtin_shufflevector(*(const v8h*)&sW[(ct * 16 + lr) * Cpad + kb * 32 + (hi ? 8 : 0)], *(const v8h*)&sW[(ct * 16 + lr) * Cpad + kb * 32 + 16 + (hi ? 8 : 0)], 0,1,2,3,4,5,6,7,8,9,10,11,12,13,14,15);
    }

    for (int rt = blockIdx.x * 8 + wave; rt < rowTiles; rt += gridDim.x * 8) {
        const h16* arow = A + (size_t)(rt * 16 + lr) * Cpad + (hi ? 8 : 0);
        v16h afrag[KB];
#pragma unroll
        for (int kb = 0; kb < KB; ++kb) {
            v8h alo = *(const v8h*)(arow + kb * 32);
            v8h ahi = *(const v8h*)(arow + kb * 32 + 16);
#pragma unroll
            for (int e = 0; e < 8; ++e) { afrag[kb][e] = alo[e]; afrag[kb][8+e] = ahi[e]; }
        }
        v8f acc[CT];
#pragma unroll
        for (int ct = 0; ct < CT; ++ct) {
#pragma unroll
            for (int j = 0; j < 8; ++j) acc[ct][j] = bv[ct];   // bias folded into C
        }
#pragma unroll
        for (int kb = 0; kb < KB; ++kb)
#pragma unroll
            for (int ct = 0; ct < CT; ++ct)
                acc[ct] = __builtin_amdgcn_wmma_f32_16x16x32_f16(false, afrag[kb], false,
                                                                 bfrag[ct][kb], (short)0,
                                                                 acc[ct], false, false);
        const int m0 = rt * 16 + (hi ? 8 : 0);
#pragma unroll
        for (int ct = 0; ct < CT; ++ct) {
            const int n = ct * 16 + lr;
#pragma unroll
            for (int j = 0; j < 8; ++j)
                H[(size_t)(m0 + j) * ldH + n] = acc[ct][j];
        }
    }
}

// =====================================================================
// 5) BatchNorm statistics: LDS-atomic partial sums, one global atomic
//    per channel per block.
// =====================================================================
__global__ __launch_bounds__(256) void reduce_stats_kernel(const float* __restrict__ H, int O,
                                                           float* __restrict__ sums,
                                                           float* __restrict__ sumsq) {
    __shared__ float ls[128], lss[128];
    const int tid = threadIdx.x;
    if (tid < 128) { ls[tid] = 0.f; lss[tid] = 0.f; }
    __syncthreads();
    const size_t total = (size_t)RROWS * O;
    for (size_t i = (size_t)blockIdx.x * blockDim.x + tid; i < total;
         i += (size_t)gridDim.x * blockDim.x) {
        float v = H[i];
        int c = (int)(i & (size_t)(O - 1));
        atomicAdd(&ls[c], v);
        atomicAdd(&lss[c], v * v);
    }
    __syncthreads();
    if (tid < O) { atomicAdd(&sums[tid], ls[tid]); atomicAdd(&sumsq[tid], lss[tid]); }
}

__global__ void finalize_stats_kernel(const float* sums, const float* sumsq,
                                      const float* gamma, const float* beta, int O,
                                      float* scale, float* shift) {
    int c = threadIdx.x;
    if (c >= O) return;
    const float inv = 1.0f / (float)RROWS;
    float mean = sums[c] * inv;
    float var  = sumsq[c] * inv - mean * mean;
    float sc = gamma[c] * rsqrtf(var + BN_EPS);
    scale[c] = sc;
    shift[c] = beta[c] - mean * sc;
}

__global__ void normalize_relu_kernel(const float* __restrict__ H, int O,
                                      const float* __restrict__ scale,
                                      const float* __restrict__ shift,
                                      h16* __restrict__ out) {
    const size_t total = (size_t)RROWS * O;
    for (size_t i = (size_t)blockIdx.x * blockDim.x + threadIdx.x; i < total;
         i += (size_t)gridDim.x * blockDim.x) {
        int c = (int)(i & (size_t)(O - 1));
        out[i] = (h16)fmaxf(H[i] * scale[c] + shift[c], 0.f);
    }
}

// last layer: normalize + ReLU fused with max over the 32 group samples
__global__ void normalize_maxpool_kernel(const float* __restrict__ H,
                                         const float* __restrict__ scale,
                                         const float* __restrict__ shift,
                                         float* __restrict__ out) {
    int t = blockIdx.x * blockDim.x + threadIdx.x;
    if (t >= BB * NPT * 128) return;
    const int g = t >> 7, o = t & 127;
    const float sc = scale[o], sh = shift[o];
    float m = -3.402823466e38f;
    const float* hp = H + (size_t)g * NSAMP * 128 + o;
    for (int k = 0; k < NSAMP; ++k) {
        float v = fmaxf(hp[(size_t)k * 128] * sc + sh, 0.f);
        m = fmaxf(m, v);
    }
    out[t] = m;
}

__global__ void zero_f_kernel(float* p, int n) {
    int t = blockIdx.x * blockDim.x + threadIdx.x;
    if (t < n) p[t] = 0.f;
}

// =====================================================================
extern "C" void kernel_launch(void* const* d_in, const int* in_sizes, int n_in,
                              void* d_out, int out_size, void* d_ws, size_t ws_size,
                              hipStream_t stream) {
    const float* xyz    = (const float*)d_in[0];
    const float* points = (const float*)d_in[1];
    const float* w0 = (const float*)d_in[2];  const float* b0 = (const float*)d_in[3];
    const float* g0 = (const float*)d_in[4];  const float* be0 = (const float*)d_in[5];
    const float* w1 = (const float*)d_in[6];  const float* b1 = (const float*)d_in[7];
    const float* g1 = (const float*)d_in[8];  const float* be1 = (const float*)d_in[9];
    const float* w2 = (const float*)d_in[10]; const float* b2 = (const float*)d_in[11];
    const float* g2 = (const float*)d_in[12]; const float* be2 = (const float*)d_in[13];
    float* out = (float*)d_out;

    // ---- workspace carve-up (256B aligned) ----
    char* ws = (char*)d_ws;
    size_t off = 0;
    auto carve = [&](size_t bytes) -> char* {
        char* p = ws + off;
        off = (off + bytes + 255) & ~(size_t)255;
        return p;
    };
    int*   idxbuf = (int*)  carve((size_t)RROWS * 4);
    h16*   featA  = (h16*)  carve((size_t)RROWS * 96 * 2);  // layer0 in; reused as layer1 out (stride 64)
    h16*   featB  = (h16*)  carve((size_t)RROWS * 64 * 2);  // layer0 out / layer1 in
    float* H      = (float*)carve((size_t)RROWS * 128 * 4); // pre-BN activations
    h16*   wh0    = (h16*)  carve(64  * 96 * 2);
    h16*   wh1    = (h16*)  carve(64  * 64 * 2);
    h16*   wh2    = (h16*)  carve(128 * 64 * 2);
    float* stats  = (float*)carve(256 * 4);                 // [0:128) sum, [128:256) sumsq
    float* scale  = (float*)carve(128 * 4);
    float* shift  = (float*)carve(128 * 4);
    float* sums = stats, *sumsq = stats + 128;

    float* newxyz = out;                    // (16,1024,3)
    float* newpts = out + BB * NPT * 3;     // (16,1024,128)

    // weights -> f16 padded
    prep_w_kernel<<<(64*96 + 255)/256, 256, 0, stream>>>(w0, wh0, 64, 67, 96);
    prep_w_kernel<<<(64*64 + 255)/256, 256, 0, stream>>>(w1, wh1, 64, 64, 64);
    prep_w_kernel<<<(128*64 + 255)/256, 256, 0, stream>>>(w2, wh2, 128, 64, 64);

    // sampling / grouping
    fps_kernel<<<BB, 1024, 0, stream>>>(xyz, newxyz);
    ballquery_kernel<<<BB, 1024, 0, stream>>>(xyz, newxyz, idxbuf);
    build_feat_kernel<<<RROWS/256, 256, 0, stream>>>(xyz, points, newxyz, idxbuf, featA);

    const int rowTiles = RROWS / 16;           // 32768 strips of 16 rows
    const int gemmGrid = 1024;                 // 8 waves/block, grid-stride over strips

    // ---- layer 0: (RROWS x 96) @ (96 x 64) ----
    gemm_wmma_kernel<4,3><<<gemmGrid, 256, 0, stream>>>(featA, wh0, b0, H, 64, rowTiles);
    zero_f_kernel<<<1, 256, 0, stream>>>(stats, 256);
    reduce_stats_kernel<<<2048, 256, 0, stream>>>(H, 64, sums, sumsq);
    finalize_stats_kernel<<<1, 128, 0, stream>>>(sums, sumsq, g0, be0, 64, scale, shift);
    normalize_relu_kernel<<<4096, 256, 0, stream>>>(H, 64, scale, shift, featB);

    // ---- layer 1: (RROWS x 64) @ (64 x 64) ----
    gemm_wmma_kernel<4,2><<<gemmGrid, 256, 0, stream>>>(featB, wh1, b1, H, 64, rowTiles);
    zero_f_kernel<<<1, 256, 0, stream>>>(stats, 256);
    reduce_stats_kernel<<<2048, 256, 0, stream>>>(H, 64, sums, sumsq);
    finalize_stats_kernel<<<1, 128, 0, stream>>>(sums, sumsq, g1, be1, 64, scale, shift);
    normalize_relu_kernel<<<4096, 256, 0, stream>>>(H, 64, scale, shift, (h16*)featA);

    // ---- layer 2: (RROWS x 64) @ (64 x 128), split into two 64-col halves ----
    gemm_wmma_kernel<4,2><<<gemmGrid, 256, 0, stream>>>((h16*)featA, wh2,         b2,      H,      128, rowTiles);
    gemm_wmma_kernel<4,2><<<gemmGrid, 256, 0, stream>>>((h16*)featA, wh2 + 64*64, b2 + 64, H + 64, 128, rowTiles);
    zero_f_kernel<<<1, 256, 0, stream>>>(stats, 256);
    reduce_stats_kernel<<<2048, 256, 0, stream>>>(H, 128, sums, sumsq);
    finalize_stats_kernel<<<1, 128, 0, stream>>>(sums, sumsq, g2, be2, 128, scale, shift);
    normalize_maxpool_kernel<<<(BB*NPT*128)/256, 256, 0, stream>>>(H, scale, shift, newpts);
}
